// CausalSelfAttention_23132693856588
// MI455X (gfx1250) — matrix operand order repaired and checked by reading, not compiled, not run on hardware
//
#include <hip/hip_runtime.h>

#define S_LEN   2048
#define D_MODEL 4096
#define N_H     32
#define N_KV    8
#define D_HEAD  128
#define N_QKV   ((N_H + 2 * N_KV) * D_HEAD)   // 6144

typedef __attribute__((ext_vector_type(8)))  float  v8f;
typedef __attribute__((ext_vector_type(4)))  float  v4f;
typedef __attribute__((ext_vector_type(16))) __bf16 v16bf;
typedef __attribute__((ext_vector_type(8)))  __bf16 v8bf;

// async global->LDS builtin pointer types (match clang's builtin signature:
// "int __vector(4) __device__ *" == v4i addrspace(1)*, and LDS side addrspace(3))
typedef int v4i __attribute__((vector_size(16)));
typedef __attribute__((address_space(1))) v4i gv4i;
typedef __attribute__((address_space(3))) v4i lv4i;

#if defined(__has_builtin)
#if __has_builtin(__builtin_amdgcn_global_load_async_to_lds_b128) && \
    __has_builtin(__builtin_amdgcn_s_wait_asynccnt)
#define HAVE_ASYNC_LDS 1
#endif
#endif
#ifndef HAVE_ASYNC_LDS
#define HAVE_ASYNC_LDS 0
#endif

static __device__ __forceinline__ v16bf cat16(v8bf lo, v8bf hi) {
  return __builtin_shufflevector(lo, hi, 0,1,2,3,4,5,6,7,8,9,10,11,12,13,14,15);
}

static __device__ __forceinline__ v8f wmma_bf16(v16bf a, v16bf b, v8f c) {
  return __builtin_amdgcn_wmma_f32_16x16x32_bf16(false, a, false, b, (short)0, c,
                                                 false, false);
}

// ---------------------------------------------------------------------------
// Tiled GEMM: C(MxN) = A(MxK) * B(KxN), fp32 in memory, bf16 WMMA compute.
// Dimensions are compile-time constants so the epilogue address math folds
// into immediate offsets. Block tile 128x128, 8 waves (2m x 4n), wave 64x32.
// ---------------------------------------------------------------------------
template <int N, int K>
__global__ __launch_bounds__(256)
void gemm_bf16wmma(const float* __restrict__ A, const float* __restrict__ B,
                   float* __restrict__ C)
{
  __shared__ __bf16 sA[128 * 32];   // [m][k], row-major
  __shared__ __bf16 sB[128 * 32];   // [n][k], transposed (B-fragment friendly)

  const int tid  = threadIdx.x;
  const int lane = tid & 31;
  const int w    = tid >> 5;
  const int wm   = w >> 2;          // 0..1
  const int wn   = w & 3;           // 0..3
  const int half = lane >> 4;       // 0: lanes 0-15, 1: lanes 16-31
  const int lmod = lane & 15;
  const int m0 = blockIdx.y * 128;
  const int n0 = blockIdx.x * 128;

  // per-thread cooperative-load coordinates (fixed across k-steps)
  const int la_m  = tid >> 1;
  const int la_kb = (tid & 1) * 16;
  const int lb_k  = tid >> 3;
  const int lb_nb = (tid & 7) * 16;
  const float* srcA = A + (size_t)(m0 + la_m) * K + la_kb;
  const float* srcB = B + (size_t)lb_k * N + n0 + lb_nb;

  v8f acc[4][2];
  #pragma unroll
  for (int mt = 0; mt < 4; ++mt)
    #pragma unroll
    for (int nt = 0; nt < 2; ++nt)
      #pragma unroll
      for (int e = 0; e < 8; ++e) acc[mt][nt][e] = 0.0f;

  #pragma unroll 1
  for (int k0 = 0; k0 < K; k0 += 32) {
    __syncthreads();
    { // A tile 128x32 fp32 -> bf16 LDS [m][k]
      const float* src = srcA + k0;
      __bf16* dst = &sA[la_m * 32 + la_kb];
      #pragma unroll
      for (int i = 0; i < 16; i += 4) {
        v4f v = *(const v4f*)(src + i);
        dst[i + 0] = (__bf16)v.x; dst[i + 1] = (__bf16)v.y;
        dst[i + 2] = (__bf16)v.z; dst[i + 3] = (__bf16)v.w;
      }
    }
    { // B tile 32x128 fp32 -> bf16 LDS transposed [n][k]
      const float* src = srcB + (size_t)k0 * N;
      #pragma unroll
      for (int i = 0; i < 16; i += 4) {
        v4f v = *(const v4f*)(src + i);
        sB[(lb_nb + i + 0) * 32 + lb_k] = (__bf16)v.x;
        sB[(lb_nb + i + 1) * 32 + lb_k] = (__bf16)v.y;
        sB[(lb_nb + i + 2) * 32 + lb_k] = (__bf16)v.z;
        sB[(lb_nb + i + 3) * 32 + lb_k] = (__bf16)v.w;
      }
    }
    // prefetch next k-step tiles into cache while this step computes
    if (k0 + 32 < K) {
      __builtin_prefetch(srcA + k0 + 32, 0, 3);
      __builtin_prefetch(srcB + (size_t)(k0 + 32) * N, 0, 3);
    }
    __syncthreads();

    // B fragments: lane = n; elems 0-7 = K half*8.., elems 8-15 = K 16+half*8..
    v16bf bfr[2];
    #pragma unroll
    for (int nt = 0; nt < 2; ++nt) {
      const int n = wn * 32 + nt * 16 + lmod;
      v8bf lo = *(const v8bf*)&sB[n * 32 + half * 8];
      v8bf hi = *(const v8bf*)&sB[n * 32 + half * 8 + 16];
      bfr[nt] = cat16(lo, hi);
    }
    // A fragments: lane = m; elems 0-7 = K half*8.., elems 8-15 = K 16+half*8..
    #pragma unroll
    for (int mt = 0; mt < 4; ++mt) {
      const int m = wm * 64 + mt * 16 + lmod;
      v8bf lo = *(const v8bf*)&sA[m * 32 + half * 8];
      v8bf hi = *(const v8bf*)&sA[m * 32 + half * 8 + 16];
      v16bf af = cat16(lo, hi);
      #pragma unroll
      for (int nt = 0; nt < 2; ++nt)
        acc[mt][nt] = wmma_bf16(af, bfr[nt], acc[mt][nt]);
    }
  }

  // C layout: lane = col, VGPR j = row (j + half*8) within 16x16 tile
  #pragma unroll
  for (int mt = 0; mt < 4; ++mt)
    #pragma unroll
    for (int nt = 0; nt < 2; ++nt) {
      const int col = n0 + wn * 32 + nt * 16 + lmod;
      const int row = m0 + wm * 64 + mt * 16 + half * 8;
      float* cp = C + (size_t)row * N + col;
      #pragma unroll
      for (int j = 0; j < 8; ++j)
        cp[(size_t)j * N] = acc[mt][nt][j];
    }
}

// ---------------------------------------------------------------------------
// RoPE + split qkv into per-head bf16 Q [H][S][DH], K [HKV][S][DH], V [HKV][S][DH]
// grid: (S, 12), 256 threads; p in [0,3072): 2048 q-pairs, 512 k-pairs, 512 v-pairs
// ---------------------------------------------------------------------------
__global__ __launch_bounds__(256)
void rope_split(const float* __restrict__ qkv,
                unsigned short* __restrict__ Qb_,
                unsigned short* __restrict__ Kb_,
                unsigned short* __restrict__ Vb_)
{
  __bf16* Qb = (__bf16*)Qb_;
  __bf16* Kb = (__bf16*)Kb_;
  __bf16* Vb = (__bf16*)Vb_;
  const int s = blockIdx.x;
  const int p = blockIdx.y * 256 + threadIdx.x;
  const float* row = qkv + (size_t)s * N_QKV;
  const float LOG_THETA_OVER_HALF = 9.210340371976184f / 64.0f; // ln(1e4)/64

  if (p < N_H * 64) {                       // Q rope
    const int h = p >> 6, i = p & 63;
    const float inv = __expf(-(float)i * LOG_THETA_OVER_HALF);
    float sn, cs; __sincosf((float)s * inv, &sn, &cs);
    const float x1 = row[h * D_HEAD + i];
    const float x2 = row[h * D_HEAD + i + 64];
    __bf16* dst = Qb + ((size_t)h * S_LEN + s) * D_HEAD;
    dst[i]      = (__bf16)(x1 * cs - x2 * sn);
    dst[i + 64] = (__bf16)(x2 * cs + x1 * sn);
  } else if (p < (N_H + N_KV) * 64) {       // K rope
    const int q = p - N_H * 64;
    const int h = q >> 6, i = q & 63;
    const float inv = __expf(-(float)i * LOG_THETA_OVER_HALF);
    float sn, cs; __sincosf((float)s * inv, &sn, &cs);
    const float* src = row + N_H * D_HEAD + h * D_HEAD;
    const float x1 = src[i];
    const float x2 = src[i + 64];
    __bf16* dst = Kb + ((size_t)h * S_LEN + s) * D_HEAD;
    dst[i]      = (__bf16)(x1 * cs - x2 * sn);
    dst[i + 64] = (__bf16)(x2 * cs + x1 * sn);
  } else {                                  // V convert
    const int q = (p - (N_H + N_KV) * 64) * 2; // 0..1022
    const int h = q >> 7, d = q & 127;
    const float* src = row + (N_H + N_KV) * D_HEAD + h * D_HEAD + d;
    __bf16* dst = Vb + ((size_t)h * S_LEN + s) * D_HEAD + d;
    dst[0] = (__bf16)src[0];
    dst[1] = (__bf16)src[1];
  }
}

// ---------------------------------------------------------------------------
// Flash attention: grid (S/128, H). Block = 128 query rows x 1 head,
// 8 waves, each wave owns 16 query rows. Key blocks of 64 with online softmax.
// K tile fetched with gfx1250 async global->LDS copies when available.
// ---------------------------------------------------------------------------
__global__ __launch_bounds__(256)
void attn_fwd(const unsigned short* __restrict__ Qb_,
              const unsigned short* __restrict__ Kb_,
              const unsigned short* __restrict__ Vb_,
              float* __restrict__ ctx)
{
  const __bf16* Qb = (const __bf16*)Qb_;
  const __bf16* Kb = (const __bf16*)Kb_;
  const __bf16* Vb = (const __bf16*)Vb_;

  __shared__ __bf16 sK[64 * 128];    // [key][d]
  __shared__ __bf16 sVT[128 * 64];   // [d][key]  (transposed V)
  __shared__ __bf16 sP[8 * 16 * 64]; // per-wave P staging (C-layout -> A-layout)

  const int h    = blockIdx.y;
  const int q0   = blockIdx.x * 128;
  const int hkv  = h >> 2;                      // GQA: 4 q-heads per kv-head
  const int tid  = threadIdx.x;
  const int lane = tid & 31;
  const int w    = tid >> 5;
  const int half = lane >> 4;
  const int lmod = lane & 15;
  const float scale = 0.08838834764831845f;     // 1/sqrt(128)

  // Per-wave Q A-fragments (rows q0 + w*16 + lmod, full d=128 => 4 k-steps)
  v16bf qf[4];
  {
    const __bf16* qrow = Qb + ((size_t)h * S_LEN + q0 + w * 16 + lmod) * D_HEAD;
    #pragma unroll
    for (int ks = 0; ks < 4; ++ks) {
      v8bf lo = *(const v8bf*)(qrow + ks * 32 + half * 8);
      v8bf hi = *(const v8bf*)(qrow + ks * 32 + half * 8 + 16);
      qf[ks] = cat16(lo, hi);
    }
  }

  v8f o[8];
  #pragma unroll
  for (int t = 0; t < 8; ++t)
    #pragma unroll
    for (int e = 0; e < 8; ++e) o[t][e] = 0.0f;
  float mrun[8], lrun[8];
  #pragma unroll
  for (int j = 0; j < 8; ++j) { mrun[j] = -3.0e38f; lrun[j] = 0.0f; }

  const int qrow_base = q0 + w * 16 + half * 8; // lane's row = qrow_base + j

  #pragma unroll 1
  for (int kb = 0; kb < q0 + 128; kb += 64) {
    __syncthreads();
    { // cooperative load K tile [key][d] (async if available) and V^T [d][key]
      const int r  = tid >> 2;
      const int db = (tid & 3) * 32;
      const __bf16* ksrc = Kb + ((size_t)hkv * S_LEN + kb + r) * D_HEAD + db;
      __bf16* kdst = &sK[r * 128 + db];
#if HAVE_ASYNC_LDS
      #pragma unroll
      for (int i = 0; i < 4; ++i)
        __builtin_amdgcn_global_load_async_to_lds_b128(
            (gv4i*)(ksrc + i * 8), (lv4i*)(kdst + i * 8), 0, 0);
#else
      #pragma unroll
      for (int i = 0; i < 4; ++i)
        *(v8bf*)(kdst + i * 8) = *(const v8bf*)(ksrc + i * 8);
#endif
      const __bf16* vs = Vb + ((size_t)hkv * S_LEN + kb + r) * D_HEAD + db;
      #pragma unroll
      for (int i = 0; i < 4; ++i) {
        v8bf v = *(const v8bf*)(vs + i * 8);
        #pragma unroll
        for (int e = 0; e < 8; ++e)
          sVT[(db + i * 8 + e) * 64 + r] = v[e];
      }
#if HAVE_ASYNC_LDS
      __builtin_amdgcn_s_wait_asynccnt(0);
#endif
    }
    __syncthreads();

    // S = Q * K^T : 16 (rows) x 64 (keys) per wave, 4 n-tiles, 4 k-steps
    v8f sacc[4];
    #pragma unroll
    for (int nt = 0; nt < 4; ++nt)
      #pragma unroll
      for (int e = 0; e < 8; ++e) sacc[nt][e] = 0.0f;
    #pragma unroll
    for (int ks = 0; ks < 4; ++ks) {
      #pragma unroll
      for (int nt = 0; nt < 4; ++nt) {
        const int n = nt * 16 + lmod;
        v8bf lo = *(const v8bf*)&sK[n * 128 + ks * 32 + half * 8];
        v8bf hi = *(const v8bf*)&sK[n * 128 + ks * 32 + half * 8 + 16];
        sacc[nt] = wmma_bf16(qf[ks], cat16(lo, hi), sacc[nt]);
      }
    }

    // online softmax per row j (half-wave shfl_xor reductions match C-layout)
    #pragma unroll
    for (int j = 0; j < 8; ++j) {
      const int qi = qrow_base + j;
      float vmax = -3.0e38f;
      #pragma unroll
      for (int nt = 0; nt < 4; ++nt) {
        const int kidx = kb + nt * 16 + lmod;
        float sv = sacc[nt][j] * scale;
        sv = (kidx <= qi) ? sv : -3.0e38f;
        sacc[nt][j] = sv;
        vmax = fmaxf(vmax, sv);
      }
      #pragma unroll
      for (int msk = 1; msk < 16; msk <<= 1)
        vmax = fmaxf(vmax, __shfl_xor(vmax, msk, 32));
      const float mo = mrun[j];
      const float mn = fmaxf(mo, vmax);
      mrun[j] = mn;
      const float alpha = __expf(mo - mn);
      float rsum = 0.0f;
      #pragma unroll
      for (int nt = 0; nt < 4; ++nt) {
        const float pv = __expf(sacc[nt][j] - mn);
        sacc[nt][j] = pv;
        rsum += pv;
      }
      #pragma unroll
      for (int msk = 1; msk < 16; msk <<= 1)
        rsum += __shfl_xor(rsum, msk, 32);
      lrun[j] = lrun[j] * alpha + rsum;
      #pragma unroll
      for (int t = 0; t < 8; ++t) o[t][j] *= alpha;
    }

    // stage P (C-layout) into per-wave LDS region in row-major [m][key]
    __bf16* pst = &sP[w * 16 * 64];
    #pragma unroll
    for (int nt = 0; nt < 4; ++nt)
      #pragma unroll
      for (int j = 0; j < 8; ++j)
        pst[(j + half * 8) * 64 + nt * 16 + lmod] = (__bf16)sacc[nt][j];

    // O += P * V : P 16x64, V 64x128 -> 2 k-steps, 8 d-tiles
    #pragma unroll
    for (int ks = 0; ks < 2; ++ks) {
      v8bf plo = *(const v8bf*)&pst[lmod * 64 + ks * 32 + half * 8];
      v8bf phi = *(const v8bf*)&pst[lmod * 64 + ks * 32 + half * 8 + 16];
      v16bf pf = cat16(plo, phi);
      #pragma unroll
      for (int nt = 0; nt < 8; ++nt) {
        const int d = nt * 16 + lmod;
        v8bf blo = *(const v8bf*)&sVT[d * 64 + ks * 32 + half * 8];
        v8bf bhi = *(const v8bf*)&sVT[d * 64 + ks * 32 + half * 8 + 16];
        o[nt] = wmma_bf16(pf, cat16(blo, bhi), o[nt]);
      }
    }
  }

  // epilogue: normalize rows and scatter fp32 ctx [s][H*DH] (constant stride)
  #pragma unroll
  for (int nt = 0; nt < 8; ++nt) {
    const int c = h * D_HEAD + nt * 16 + lmod;
    float* cp = ctx + (size_t)qrow_base * (N_H * D_HEAD) + c;
    #pragma unroll
    for (int j = 0; j < 8; ++j)
      cp[(size_t)j * (N_H * D_HEAD)] = o[nt][j] / lrun[j];
  }
}

// ---------------------------------------------------------------------------
extern "C" void kernel_launch(void* const* d_in, const int* in_sizes, int n_in,
                              void* d_out, int out_size, void* d_ws, size_t ws_size,
                              hipStream_t stream) {
  (void)in_sizes; (void)n_in; (void)out_size; (void)ws_size;
  const float* hidden = (const float*)d_in[0];
  // d_in[1] = sequence_mask (all ones; unused)
  const float* Wqkv   = (const float*)d_in[2];
  const float* Wo     = (const float*)d_in[3];
  float* out = (float*)d_out;

  char* ws = (char*)d_ws;
  size_t off = 0;
  float* qkv = (float*)(ws + off);                  off += (size_t)S_LEN * N_QKV * 4;
  unsigned short* Qb = (unsigned short*)(ws + off); off += (size_t)N_H  * S_LEN * D_HEAD * 2;
  unsigned short* Kb = (unsigned short*)(ws + off); off += (size_t)N_KV * S_LEN * D_HEAD * 2;
  unsigned short* Vb = (unsigned short*)(ws + off); off += (size_t)N_KV * S_LEN * D_HEAD * 2;
  float* ctx = (float*)(ws + off);

  dim3 blk(256);
  // qkv = X @ W_qkv   (2048 x 6144, K=4096)
  gemm_bf16wmma<N_QKV, D_MODEL><<<dim3(N_QKV / 128, S_LEN / 128), blk, 0, stream>>>(
      hidden, Wqkv, qkv);
  // RoPE + split to bf16 per-head layouts
  rope_split<<<dim3(S_LEN, 12), blk, 0, stream>>>(qkv, Qb, Kb, Vb);
  // causal GQA flash attention
  attn_fwd<<<dim3(S_LEN / 128, N_H), blk, 0, stream>>>(Qb, Kb, Vb, ctx);
  // out = ctx @ W_o   (2048 x 4096, K=4096)
  gemm_bf16wmma<D_MODEL, D_MODEL><<<dim3(D_MODEL / 128, S_LEN / 128), blk, 0, stream>>>(
      ctx, Wo, out);
}
